// FusedMoE_20444044329637
// MI455X (gfx1250) — matrix operand order repaired and checked by reading, not compiled, not run on hardware
//
#include <hip/hip_runtime.h>

// ---------------- problem constants (match reference) ----------------
#define E_EXPERTS 16
#define KDIM      1536   // hidden size
#define NDIM      1536   // intermediate size
#define TTOK      4096   // tokens
#define TOPK      2
#define MT        64     // tokens per expert tile (M block)
#define LDA       1544   // padded LDS row stride in bf16 elems (1544*2/4 mod 64 = 4 -> conflict free)

typedef __attribute__((ext_vector_type(16))) __bf16 v16bf;
typedef __attribute__((ext_vector_type(8)))  __bf16 v8bf;
typedef __attribute__((ext_vector_type(8)))  float  v8f;
typedef __attribute__((ext_vector_type(4)))  float  v4f;

// ---------------- WMMA helpers ----------------
__device__ __forceinline__ v8f wmma_bf16(v16bf a, v16bf b, v8f c) {
    // D = A(16x32 bf16) x B(32x16 bf16) + C(16x16 f32)
    return __builtin_amdgcn_wmma_f32_16x16x32_bf16(false, a, false, b, (short)0, c, false, false);
}

// B fragment: lane holds column n = lane&15, half = lane>>4.
// Element i is B[k][n], k = kb + half*16 + i; B[k][n] = W[n_row][k] (row-major, dot over K)
// -> one contiguous 16-float (64B) run per lane; half-pairs cover a full 128B line.
__device__ __forceinline__ v16bf load_b_f32(const float* __restrict__ p) {
    v4f a = *(const v4f*)(p);
    v4f b = *(const v4f*)(p + 4);
    v4f c = *(const v4f*)(p + 8);
    v4f d = *(const v4f*)(p + 12);
    v16bf r;
#pragma unroll
    for (int i = 0; i < 4; ++i) {
        r[i]      = (__bf16)a[i];
        r[4 + i]  = (__bf16)b[i];
        r[8 + i]  = (__bf16)c[i];
        r[12 + i] = (__bf16)d[i];
    }
    return r;
}

// A fragment from LDS: lane holds row m = lane&15, half = lane>>4.
// Elements 0..7  -> K = kb + half*16 + 0..7
// Elements 8..15 -> K = kb + half*16 + 8 + 0..7   (two 16B ds_load_b128)
__device__ __forceinline__ v16bf load_a_lds(const __bf16* __restrict__ arow, int kb, int half) {
    union { v16bf v; v8bf h[2]; } u;
    u.h[0] = *(const v8bf*)(arow + kb + half * 16);
    u.h[1] = *(const v8bf*)(arow + kb + half * 16 + 8);
    return u.v;
}

// ---------------- kernel 0: zero output + counters ----------------
__global__ void moe_zero(float* __restrict__ out, int* __restrict__ flags) {
    int tid = blockIdx.x * blockDim.x + threadIdx.x;
    if (tid < 32) flags[tid] = 0;  // counts[16] + cursor[16]
    v4f z = {0.f, 0.f, 0.f, 0.f};
    size_t n4 = (size_t)TTOK * KDIM / 4;
    for (size_t i = tid; i < n4; i += (size_t)gridDim.x * blockDim.x)
        ((v4f*)out)[i] = z;
}

// ---------------- kernel 1: router (softmax + top-2 renormalized) ----------------
__global__ void moe_router(const float* __restrict__ logits, int* __restrict__ counts,
                           int* __restrict__ topk_id, float* __restrict__ topk_w) {
    int t = blockIdx.x * blockDim.x + threadIdx.x;
    if (t >= TTOK) return;
    const float* l = logits + (size_t)t * E_EXPERTS;
    float b1 = -3.4e38f, b2 = -3.4e38f;
    int i1 = 0, i2 = 0;
#pragma unroll
    for (int i = 0; i < E_EXPERTS; ++i) {
        float v = l[i];
        if (v > b1)      { b2 = b1; i2 = i1; b1 = v; i1 = i; }
        else if (v > b2) { b2 = v; i2 = i; }
    }
    // renormalized top-2 softmax: denominator cancels
    float e2 = __expf(b2 - b1);
    float inv = 1.f / (1.f + e2);
    topk_id[2 * t]     = i1;
    topk_id[2 * t + 1] = i2;
    topk_w[2 * t]      = inv;
    topk_w[2 * t + 1]  = e2 * inv;
    atomicAdd(&counts[i1], 1);
    atomicAdd(&counts[i2], 1);
}

// ---------------- kernel 2: exclusive scan over 16 experts ----------------
__global__ void moe_scan(const int* __restrict__ counts, int* __restrict__ offsets) {
    if (threadIdx.x == 0 && blockIdx.x == 0) {
        int s = 0;
        for (int i = 0; i < E_EXPERTS; ++i) { offsets[i] = s; s += counts[i]; }
        offsets[E_EXPERTS] = s;
    }
}

// ---------------- kernel 3: scatter assignments grouped by expert ----------------
__global__ void moe_scatter(const int* __restrict__ topk_id, const float* __restrict__ topk_w,
                            const int* __restrict__ offsets, int* __restrict__ cursor,
                            int* __restrict__ assign_tok, float* __restrict__ assign_w) {
    int i = blockIdx.x * blockDim.x + threadIdx.x;
    if (i >= TTOK * TOPK) return;
    int e = topk_id[i];
    int pos = offsets[e] + atomicAdd(&cursor[e], 1);
    assign_tok[pos] = i >> 1;
    assign_w[pos] = topk_w[i];
}

// ---------------- kernel 4: H = silu(x W13g^T + bg) * (x W13u^T + bu) ----------------
// grid (tile, expert): consecutive blocks share one expert's weights -> L2 resident.
// Block = 256 thr = 8 waves; wave tile = 16 rows x 64 cols.
__global__ __launch_bounds__(256) void moe_w13(
    const float* __restrict__ x, const float* __restrict__ w13, const float* __restrict__ b13,
    const int* __restrict__ offsets, const int* __restrict__ assign_tok,
    __bf16* __restrict__ Hbuf) {
    extern __shared__ char smem[];
    __bf16* As = (__bf16*)smem;                       // [MT][LDA] bf16 activations
    int* toks = (int*)(smem + (size_t)MT * LDA * 2);  // [MT]

    int e = blockIdx.y;
    int beg = offsets[e], end_ = offsets[e + 1];
    int base = beg + blockIdx.x * MT;
    if (base >= end_) return;
    int rows = end_ - base; if (rows > MT) rows = MT;
    int tid = threadIdx.x;

    if (tid < MT) toks[tid] = assign_tok[base + (tid < rows ? tid : 0)];
    __syncthreads();

    // stage gathered A tile: 64 x 1536, fp32 -> bf16 into LDS
    for (int i = tid; i < MT * (KDIM / 4); i += 256) {
        int r = i / (KDIM / 4), c4 = i - r * (KDIM / 4);
        v4f f = *(const v4f*)(x + (size_t)toks[r] * KDIM + c4 * 4);
        __bf16* d = As + r * LDA + c4 * 4;
        d[0] = (__bf16)f[0]; d[1] = (__bf16)f[1];
        d[2] = (__bf16)f[2]; d[3] = (__bf16)f[3];
    }
    __syncthreads();

    int w = tid >> 5, lane = tid & 31;
    int wm = w >> 1, wn = w & 1;
    int half = lane >> 4, l16 = lane & 15;
    const __bf16* arow = As + (wm * 16 + l16) * LDA;
    const float* wbase = w13 + (size_t)e * (2 * NDIM) * KDIM;
    const v8f vzero = {0.f, 0.f, 0.f, 0.f, 0.f, 0.f, 0.f, 0.f};

    for (int nb = 0; nb < NDIM; nb += 128) {
        int ncol = nb + wn * 64 + l16;  // this lane's column within tile t4: ncol + t4*16
        v8f accG[4], accU[4];
#pragma unroll
        for (int t4 = 0; t4 < 4; ++t4) { accG[t4] = vzero; accU[t4] = vzero; }

        for (int kb = 0; kb < KDIM; kb += 32) {
            v16bf afrag = load_a_lds(arow, kb, half);
            // materialize all 8 B fragments first: distinct regs let the cvts of
            // the next step co-execute under this step's WMMAs (no WAR v_nops)
            v16bf bg[4], bu[4];
#pragma unroll
            for (int t4 = 0; t4 < 4; ++t4) {
                const float* gp = wbase + (size_t)(ncol + t4 * 16) * KDIM + kb + half * 16;
                const float* up = wbase + (size_t)(NDIM + ncol + t4 * 16) * KDIM + kb + half * 16;
                bg[t4] = load_b_f32(gp);
                bu[t4] = load_b_f32(up);
            }
#pragma unroll
            for (int t4 = 0; t4 < 4; ++t4) {
                accG[t4] = wmma_bf16(afrag, bg[t4], accG[t4]);
                accU[t4] = wmma_bf16(afrag, bu[t4], accU[t4]);
            }
        }
        // epilogue: bias + SwiGLU, store bf16 H
#pragma unroll
        for (int t4 = 0; t4 < 4; ++t4) {
            int n = ncol + t4 * 16;
            float bg = b13[(size_t)e * 2 * NDIM + n];
            float bu = b13[(size_t)e * 2 * NDIM + NDIM + n];
            int rowbase = wm * 16 + half * 8;
#pragma unroll
            for (int r = 0; r < 8; ++r) {
                int row = rowbase + r;
                float g = accG[t4][r] + bg;
                float u = accU[t4][r] + bu;
                float h = g / (1.f + __expf(-g)) * u;
                if (row < rows)
                    Hbuf[(size_t)(base + row) * NDIM + n] = (__bf16)h;
            }
        }
    }
}

// ---------------- kernel 5: out += ce * (H W2^T + b2), atomic combine ----------------
__global__ __launch_bounds__(256) void moe_w2(
    const __bf16* __restrict__ Hbuf, const float* __restrict__ w2, const float* __restrict__ b2,
    const int* __restrict__ offsets, const int* __restrict__ assign_tok,
    const float* __restrict__ assign_w, float* __restrict__ out) {
    extern __shared__ char smem[];
    __bf16* Hs = (__bf16*)smem;
    int* toks = (int*)(smem + (size_t)MT * LDA * 2);
    float* cw = (float*)(smem + (size_t)MT * LDA * 2 + MT * 4);

    int e = blockIdx.y;
    int beg = offsets[e], end_ = offsets[e + 1];
    int base = beg + blockIdx.x * MT;
    if (base >= end_) return;
    int rows = end_ - base; if (rows > MT) rows = MT;
    int tid = threadIdx.x;

    if (tid < MT) {
        int idx = tid < rows ? tid : 0;
        toks[tid] = assign_tok[base + idx];
        cw[tid] = (tid < rows) ? assign_w[base + tid] : 0.f;
    }
    __syncthreads();

    // stage H tile (already bf16) into LDS with async copies: no VGPR round-trip,
    // tracked by ASYNCcnt (CDNA5 async-to-LDS path).
    for (int i = tid; i < MT * (NDIM / 8); i += 256) {
        int r = i / (NDIM / 8), c8 = i - r * (NDIM / 8);
        int rr = r < rows ? r : 0;
        const __bf16* gsrc = Hbuf + (size_t)(base + rr) * NDIM + c8 * 8;
        // low 32 bits of the flat shared-aperture address are the LDS byte offset
        unsigned ldsOff = (unsigned)(unsigned long long)(Hs + r * LDA + c8 * 8);
        asm volatile("global_load_async_to_lds_b128 %0, %1, off"
                     :: "v"(ldsOff), "v"(gsrc) : "memory");
    }
    asm volatile("s_wait_asynccnt 0x0" ::: "memory");
    __syncthreads();

    int w = tid >> 5, lane = tid & 31;
    int wm = w >> 1, wn = w & 1;
    int half = lane >> 4, l16 = lane & 15;
    const __bf16* arow = Hs + (wm * 16 + l16) * LDA;
    const float* w2base = w2 + (size_t)e * KDIM * NDIM;
    const v8f vzero = {0.f, 0.f, 0.f, 0.f, 0.f, 0.f, 0.f, 0.f};

    for (int kc = 0; kc < KDIM; kc += 128) {
        int kcol0 = kc + wn * 64 + l16;
        v8f acc[4];
#pragma unroll
        for (int t4 = 0; t4 < 4; ++t4) acc[t4] = vzero;

        for (int nb = 0; nb < NDIM; nb += 32) {
            v16bf afrag = load_a_lds(arow, nb, half);
            v16bf bfr[4];
#pragma unroll
            for (int t4 = 0; t4 < 4; ++t4) {
                const float* bp = w2base + (size_t)(kcol0 + t4 * 16) * NDIM + nb + half * 16;
                bfr[t4] = load_b_f32(bp);
            }
#pragma unroll
            for (int t4 = 0; t4 < 4; ++t4)
                acc[t4] = wmma_bf16(afrag, bfr[t4], acc[t4]);
        }
#pragma unroll
        for (int t4 = 0; t4 < 4; ++t4) {
            int kcol = kcol0 + t4 * 16;
            float bias = b2[(size_t)e * KDIM + kcol];
            int rowbase = wm * 16 + half * 8;
#pragma unroll
            for (int r = 0; r < 8; ++r) {
                int row = rowbase + r;
                if (row < rows) {
                    float v = cw[row] * (acc[t4][r] + bias);
                    atomicAdd(out + (size_t)toks[row] * KDIM + kcol, v);
                }
            }
        }
    }
}

// ---------------- launch ----------------
extern "C" void kernel_launch(void* const* d_in, const int* in_sizes, int n_in,
                              void* d_out, int out_size, void* d_ws, size_t ws_size,
                              hipStream_t stream) {
    const float* x      = (const float*)d_in[0];  // [T,K]
    const float* logits = (const float*)d_in[1];  // [T,E]
    const float* w13    = (const float*)d_in[2];  // [E,2N,K]
    const float* b13    = (const float*)d_in[3];  // [E,2N]
    const float* w2     = (const float*)d_in[4];  // [E,K,N]
    const float* b2     = (const float*)d_in[5];  // [E,K]
    float* out = (float*)d_out;

    char* ws = (char*)d_ws;
    int*    counts     = (int*)(ws + 0);       // 16
    int*    cursor     = (int*)(ws + 64);      // 16
    int*    offsets    = (int*)(ws + 128);     // 17
    int*    topk_id    = (int*)(ws + 256);     // 8192
    float*  topk_w     = (float*)(ws + 33024); // 8192
    int*    assign_tok = (int*)(ws + 65792);   // 8192
    float*  assign_w   = (float*)(ws + 98560); // 8192
    __bf16* Hbuf       = (__bf16*)(ws + 131328); // 8192*1536 bf16 = 24 MiB

    moe_zero<<<1024, 256, 0, stream>>>(out, counts);
    moe_router<<<TTOK / 256, 256, 0, stream>>>(logits, counts, topk_id, topk_w);
    moe_scan<<<1, 1, 0, stream>>>(counts, offsets);
    moe_scatter<<<(TTOK * TOPK) / 256, 256, 0, stream>>>(topk_id, topk_w, offsets, cursor,
                                                         assign_tok, assign_w);
    size_t shmem = (size_t)MT * LDA * 2 + MT * 4 + MT * 4;  // ~194 KB (<=320 KB/WGP)
    // x = tile, y = expert: consecutive blocks reuse the same expert's weights in L2
    dim3 grid(TTOK / MT, E_EXPERTS);  // blocks past an expert's count exit early
    moe_w13<<<grid, 256, shmem, stream>>>(x, w13, b13, offsets, assign_tok, Hbuf);
    moe_w2<<<grid, 256, shmem, stream>>>(Hbuf, w2, b2, offsets, assign_tok, assign_w, out);
}
